// HiddenMarkovModel_64785286693695
// MI455X (gfx1250) — matrix operand order repaired and checked by reading, not compiled, not run on hardware
//
#include <hip/hip_runtime.h>
#include <hip/hip_bf16.h>

// ---------------------------------------------------------------------------
// HMM forward log-likelihood on gfx1250 (MI455X), wave32 + WMMA bf16.
//   B=64, T=512, N=256, D=128
// ---------------------------------------------------------------------------

typedef __attribute__((ext_vector_type(16))) __bf16 v16bf;
typedef __attribute__((ext_vector_type(8)))  __bf16 v8bf;
typedef __attribute__((ext_vector_type(8)))  float  v8f;

#define HMM_B 64
#define HMM_T 512
#define HMM_N 256
#define HMM_D 128
#define LOG_2PI 1.8378770664093453f

static __device__ __forceinline__ v16bf cat8(v8bf lo, v8bf hi) {
    return __builtin_shufflevector(lo, hi, 0,1,2,3,4,5,6,7,8,9,10,11,12,13,14,15);
}

static __device__ __forceinline__ v8f wmma_bf16(v16bf a, v16bf b, v8f c) {
    return __builtin_amdgcn_wmma_f32_16x16x32_bf16(
        /*neg_a=*/false, a, /*neg_b=*/false, b,
        /*c_mod=*/(short)0, c, /*reuse_a=*/false, /*reuse_b=*/false);
}

// ---------------------------------------------------------------------------
// prep: build
//   BpT  [n][k] bf16 : k<128 -> exp(-log_var[n][k]) ; k>=128 -> -2*mu*iv
//   cnst [n]   f32  : sum_d (mu^2*iv + log_var + LOG2PI)
//   AexpT[j][i] bf16 : row-softmax of transition, transposed (K-contiguous)
//   log_pi[n]  f32  : log_softmax(state_priors)
// grid = 513 blocks x 256 threads
// ---------------------------------------------------------------------------
__global__ void hmm_prep_kernel(const float* __restrict__ mu,
                                const float* __restrict__ log_var,
                                const float* __restrict__ trans,
                                const float* __restrict__ priors,
                                __bf16* __restrict__ BpT,
                                __bf16* __restrict__ AexpT,
                                float* __restrict__ cnst,
                                float* __restrict__ log_pi) {
    __shared__ float red[256];
    const int tid = threadIdx.x;
    const int blk = blockIdx.x;

    if (blk < HMM_N) {                       // BpT row + cnst[n]
        const int n = blk;
        float contrib = 0.f;
        if (tid < HMM_D) {
            const int d = tid;
            const float lv = log_var[n * HMM_D + d];
            const float iv = __expf(-lv);
            const float m  = mu[n * HMM_D + d];
            BpT[n * 256 + d]       = (__bf16)iv;
            BpT[n * 256 + 128 + d] = (__bf16)(-2.f * m * iv);
            contrib = m * m * iv + lv + LOG_2PI;
        }
        red[tid] = contrib;
        __syncthreads();
        for (int s = 128; s > 0; s >>= 1) {
            if (tid < s) red[tid] += red[tid + s];
            __syncthreads();
        }
        if (tid == 0) cnst[n] = red[0];
    } else if (blk < 2 * HMM_N) {            // row softmax of transition row i
        const int i = blk - HMM_N;
        const float v = trans[i * HMM_N + tid];
        red[tid] = v;
        __syncthreads();
        for (int s = 128; s > 0; s >>= 1) {
            if (tid < s) red[tid] = fmaxf(red[tid], red[tid + s]);
            __syncthreads();
        }
        const float mx = red[0];
        __syncthreads();
        const float e = __expf(v - mx);
        red[tid] = e;
        __syncthreads();
        for (int s = 128; s > 0; s >>= 1) {
            if (tid < s) red[tid] += red[tid + s];
            __syncthreads();
        }
        const float sum = red[0];
        AexpT[tid * HMM_N + i] = (__bf16)(e / sum);   // [j=tid][i] layout
    } else {                                  // log_softmax(state_priors)
        const float v = priors[tid];
        red[tid] = v;
        __syncthreads();
        for (int s = 128; s > 0; s >>= 1) {
            if (tid < s) red[tid] = fmaxf(red[tid], red[tid + s]);
            __syncthreads();
        }
        const float mx = red[0];
        __syncthreads();
        red[tid] = __expf(v - mx);
        __syncthreads();
        for (int s = 128; s > 0; s >>= 1) {
            if (tid < s) red[tid] += red[tid + s];
            __syncthreads();
        }
        log_pi[tid] = v - mx - __logf(red[0]);
    }
}

// ---------------------------------------------------------------------------
// emission: E[m][n] = -0.5 * (A'[m,:].B'[:,n] + cnst[n]),  m = b*T + t
// One wave = one 16(M)x16(N) output tile looped over all 16 N-tiles (A-frags
// built once from X, kept in VGPRs). grid = 256 blocks x 256 threads.
// ---------------------------------------------------------------------------
__global__ void hmm_emission_kernel(const float* __restrict__ X,
                                    const __bf16* __restrict__ BpT,
                                    const float* __restrict__ cnst,
                                    float* __restrict__ E) {
    const int lane = threadIdx.x & 31;
    const int w    = threadIdx.x >> 5;
    const bool hi  = lane >= 16;
    const int mrow = lane & 15;
    const int m    = blockIdx.x * 128 + w * 16 + mrow;
    const float* xr = X + (size_t)m * HMM_D;

    // Build A' fragments: A[0..3] = X^2 (K=0..127), A[4..7] = X (K=128..255).
    v16bf A[8];
#pragma unroll
    for (int kc2 = 0; kc2 < 4; ++kc2) {
        const int d0 = kc2 * 32 + (hi ? 8 : 0);
        const float4 xa = *(const float4*)(xr + d0);
        const float4 xb = *(const float4*)(xr + d0 + 4);
        const float4 xc = *(const float4*)(xr + d0 + 16);
        const float4 xd = *(const float4*)(xr + d0 + 20);
        const float xs[16] = {xa.x, xa.y, xa.z, xa.w, xb.x, xb.y, xb.z, xb.w,
                              xc.x, xc.y, xc.z, xc.w, xd.x, xd.y, xd.z, xd.w};
#pragma unroll
        for (int e = 0; e < 16; ++e) {
            A[kc2][e]     = (__bf16)(xs[e] * xs[e]);
            A[4 + kc2][e] = (__bf16)xs[e];
        }
    }

    const int mbase = blockIdx.x * 128 + w * 16 + (hi ? 8 : 0);
    for (int nt = 0; nt < 16; ++nt) {
        const int n = nt * 16 + mrow;
        const __bf16* bc = BpT + (size_t)n * 256;
        v8f acc = {};
#pragma unroll
        for (int kc = 0; kc < 8; ++kc) {
            const v8bf* bp = (const v8bf*)(bc + kc * 32 + (hi ? 8 : 0));
            acc = wmma_bf16(A[kc], cat8(bp[0], bp[2]), acc);
        }
        const float cn = cnst[n];
#pragma unroll
        for (int r = 0; r < 8; ++r)
            E[(size_t)(mbase + r) * HMM_N + n] = -0.5f * (acc[r] + cn);
    }
}

// ---------------------------------------------------------------------------
__global__ void hmm_zero_kernel(float* __restrict__ out) {
    if (threadIdx.x == 0) out[0] = 0.f;
}

// ---------------------------------------------------------------------------
// forward: 16 batches per workgroup, 511 sequential steps.
//   alpha'[m][j] = m_r + log( exp(alpha-m_r) . Aexp[:,j] ) + E[b,t,j]
// B-fragments of Aexp^T resident in VGPRs for the whole kernel.
// grid = 4 blocks x 256 threads (8 waves; wave w owns columns [32w,32w+32)).
// ---------------------------------------------------------------------------
__global__ void hmm_forward_kernel(const float* __restrict__ E,
                                   const __bf16* __restrict__ AexpT,
                                   const float* __restrict__ log_pi,
                                   float* __restrict__ out) {
    __shared__ float  s_alpha[16 * 256];
    __shared__ __align__(16) __bf16 s_P[16 * 256];
    __shared__ float  s_pm[16 * 16];
    __shared__ float  s_m[16];
    __shared__ float  s_lp[16];

    const int tid   = threadIdx.x;
    const int lane  = tid & 31;
    const int w     = tid >> 5;
    const bool hi   = lane >= 16;
    const int mrow  = lane & 15;
    const int row   = tid & 15;     // elementwise row ownership
    const int chunk = tid >> 4;     // elementwise 16-col chunk ownership
    const int b0    = blockIdx.x * 16;

    // alpha0 = log_pi + E[:,0,:]
#pragma unroll
    for (int c = 0; c < 16; ++c) {
        const int n = chunk * 16 + c;
        s_alpha[row * 256 + n] =
            log_pi[n] + E[(size_t)(b0 + row) * HMM_T * HMM_N + n];
    }

    // Constant B fragments (Aexp^T), kept in VGPRs for all 511 steps.
    v16bf Bf[2][8];
#pragma unroll
    for (int tt = 0; tt < 2; ++tt) {
        const int n = (2 * w + tt) * 16 + mrow;
        const __bf16* bc = AexpT + (size_t)n * 256;
#pragma unroll
        for (int kc = 0; kc < 8; ++kc) {
            const v8bf* bp = (const v8bf*)(bc + kc * 32 + (hi ? 8 : 0));
            Bf[tt][kc] = cat8(bp[0], bp[2]);
        }
    }
    __syncthreads();

    for (int t = 1; t < HMM_T; ++t) {
        // Prefetch this step's emission values (register-resident, hides L2).
        float epre[2][8];
#pragma unroll
        for (int tt = 0; tt < 2; ++tt) {
            const int n = (2 * w + tt) * 16 + mrow;
#pragma unroll
            for (int r = 0; r < 8; ++r) {
                const int rr = r + (hi ? 8 : 0);
                epre[tt][r] =
                    E[((size_t)(b0 + rr) * HMM_T + t) * HMM_N + n];
            }
        }

        // 1) partial row max
        float pm = -3.402823e38f;
#pragma unroll
        for (int c = 0; c < 16; ++c)
            pm = fmaxf(pm, s_alpha[row * 256 + chunk * 16 + c]);
        s_pm[row * 16 + chunk] = pm;
        __syncthreads();

        // 2) full row max
        if (tid < 16) {
            float mx = s_pm[tid * 16];
#pragma unroll
            for (int k = 1; k < 16; ++k) mx = fmaxf(mx, s_pm[tid * 16 + k]);
            s_m[tid] = mx;
        }
        __syncthreads();

        // 3) P = exp(alpha - m) in bf16 (A-matrix source)
        const float mrw = s_m[row];
#pragma unroll
        for (int c = 0; c < 16; ++c) {
            const int idx = row * 256 + chunk * 16 + c;
            s_P[idx] = (__bf16)__expf(s_alpha[idx] - mrw);
        }
        __syncthreads();

        // 4) 16x256x256 WMMA GEMM: P (16xK) x Aexp (KxN)
        v8f acc0 = {}, acc1 = {};
#pragma unroll
        for (int kc = 0; kc < 8; ++kc) {
            const v8bf* ap =
                (const v8bf*)(s_P + mrow * 256 + kc * 32 + (hi ? 8 : 0));
            const v16bf afrag = cat8(ap[0], ap[2]);   // K {0,16} / {8,24} split
            acc0 = wmma_bf16(afrag, Bf[0][kc], acc0);
            acc1 = wmma_bf16(afrag, Bf[1][kc], acc1);
        }

        // 5) epilogue: alpha' = m + log(acc) + E_t
#pragma unroll
        for (int tt = 0; tt < 2; ++tt) {
            const int n = (2 * w + tt) * 16 + mrow;
#pragma unroll
            for (int r = 0; r < 8; ++r) {
                const int rr = r + (hi ? 8 : 0);
                const float a = tt ? acc1[r] : acc0[r];
                s_alpha[rr * 256 + n] = s_m[rr] + __logf(a) + epre[tt][r];
            }
        }
        __syncthreads();
    }

    // Final: log_props[b] = logsumexp_n alpha[b][n]; accumulate sum over B.
    float pm = -3.402823e38f;
#pragma unroll
    for (int c = 0; c < 16; ++c)
        pm = fmaxf(pm, s_alpha[row * 256 + chunk * 16 + c]);
    s_pm[row * 16 + chunk] = pm;
    __syncthreads();
    if (tid < 16) {
        float mx = s_pm[tid * 16];
#pragma unroll
        for (int k = 1; k < 16; ++k) mx = fmaxf(mx, s_pm[tid * 16 + k]);
        s_m[tid] = mx;
    }
    __syncthreads();
    float ssum = 0.f;
#pragma unroll
    for (int c = 0; c < 16; ++c)
        ssum += __expf(s_alpha[row * 256 + chunk * 16 + c] - s_m[row]);
    s_pm[row * 16 + chunk] = ssum;
    __syncthreads();
    if (tid < 16) {
        float s = 0.f;
#pragma unroll
        for (int k = 0; k < 16; ++k) s += s_pm[tid * 16 + k];
        s_lp[tid] = s_m[tid] + __logf(s);
    }
    __syncthreads();
    if (tid == 0) {
        float tot = 0.f;
#pragma unroll
        for (int k = 0; k < 16; ++k) tot += s_lp[k];
        atomicAdd(out, tot);
    }
}

// ---------------------------------------------------------------------------
extern "C" void kernel_launch(void* const* d_in, const int* in_sizes, int n_in,
                              void* d_out, int out_size, void* d_ws, size_t ws_size,
                              hipStream_t stream) {
    (void)in_sizes; (void)n_in; (void)out_size; (void)ws_size;

    const float* X       = (const float*)d_in[0];   // [B,T,D]
    const float* mu      = (const float*)d_in[1];   // [N,D]
    const float* log_var = (const float*)d_in[2];   // [N,D]
    const float* trans   = (const float*)d_in[3];   // [N,N]
    const float* priors  = (const float*)d_in[4];   // [N]
    float* out = (float*)d_out;

    // Workspace layout (base is 256B-aligned):
    char* p = (char*)d_ws;
    size_t o = 0;
    float*  E      = (float*)(p + o);  o += (size_t)HMM_B * HMM_T * HMM_N * 4; // 33.5 MB
    __bf16* BpT    = (__bf16*)(p + o); o += (size_t)HMM_N * 256 * 2;            // 128 KB
    __bf16* AexpT  = (__bf16*)(p + o); o += (size_t)HMM_N * HMM_N * 2;          // 128 KB
    float*  cnst   = (float*)(p + o);  o += 1024;
    float*  lpi    = (float*)(p + o);  o += 1024;

    hmm_prep_kernel<<<dim3(2 * HMM_N + 1), dim3(256), 0, stream>>>(
        mu, log_var, trans, priors, BpT, AexpT, cnst, lpi);

    hmm_emission_kernel<<<dim3((HMM_B * HMM_T) / 128), dim3(256), 0, stream>>>(
        X, BpT, cnst, E);

    hmm_zero_kernel<<<dim3(1), dim3(32), 0, stream>>>(out);

    hmm_forward_kernel<<<dim3(HMM_B / 16), dim3(256), 0, stream>>>(
        E, AexpT, lpi, out);
}
